// NaiveGroupedGEMM_27874337751309
// MI455X (gfx1250) — matrix operand order repaired and checked by reading, not compiled, not run on hardware
//
#include <hip/hip_runtime.h>

typedef __attribute__((ext_vector_type(16))) __bf16 v16bf;
typedef __attribute__((ext_vector_type(8)))  __bf16 v8bf;
typedef __attribute__((ext_vector_type(4)))  __bf16 v4bf;
typedef __attribute__((ext_vector_type(8)))  float  v8f;

namespace {
constexpr int E  = 16;
constexpr int M  = 256;
constexpr int K  = 2048;
constexpr int N  = 8192;
constexpr int NT = 64;        // N columns per workgroup
constexpr int KB = 32;        // K per stage == bf16 WMMA depth
constexpr int KBP = KB + 8;   // padded LDS row stride (bf16 units) -> conflict-free b128 reads
}

__global__ __launch_bounds__(256, 2)
void grouped_gemm_bf16x3_wmma(const float* __restrict__ X,
                              const float* __restrict__ W,
                              float* __restrict__ O) {
  // Split-precision staging tiles: A (256 x 32) and B (64 x 32), hi/lo bf16.
  __shared__ alignas(16) __bf16 lAhi[M * KBP];
  __shared__ alignas(16) __bf16 lAlo[M * KBP];
  __shared__ alignas(16) __bf16 lBhi[NT * KBP];
  __shared__ alignas(16) __bf16 lBlo[NT * KBP];

  const int e     = blockIdx.y;
  const int nBase = blockIdx.x * NT;

  const float* Ae = X + (size_t)e * M * K;
  const float* We = W + (size_t)e * N * K + (size_t)nBase * K;  // W rows are along K
  float*       Ce = O + (size_t)e * M * N + nBase;

  const int tid  = threadIdx.x;
  const int wave = tid >> 5;
  const int lane = tid & 31;
  const int half = lane >> 4;   // 0: lanes 0-15, 1: lanes 16-31
  const int l16  = lane & 15;
  const int mWave = wave * 32;  // each wave owns 32 M rows (2 tiles) x 64 N cols (4 tiles)

  // Staging decomposition: 8 threads per row, float4 per thread (128B coalesced)
  const int sr = tid >> 3;        // 0..31
  const int sc = (tid & 7) * 4;   // 0,4,...,28

  v8f acc[2][4] = {};

  for (int k0 = 0; k0 < K; k0 += KB) {
    __syncthreads();  // protect LDS from previous iteration's readers

    // ---- Stage A block (M x KB) as split bf16 (hi + lo) ----
    #pragma unroll
    for (int r = sr; r < M; r += 32) {
      const float4 v = *(const float4*)(Ae + (size_t)r * K + (k0 + sc));
      v4bf h, l;
      h[0] = (__bf16)v.x; l[0] = (__bf16)(v.x - (float)h[0]);
      h[1] = (__bf16)v.y; l[1] = (__bf16)(v.y - (float)h[1]);
      h[2] = (__bf16)v.z; l[2] = (__bf16)(v.z - (float)h[2]);
      h[3] = (__bf16)v.w; l[3] = (__bf16)(v.w - (float)h[3]);
      *(v4bf*)&lAhi[r * KBP + sc] = h;
      *(v4bf*)&lAlo[r * KBP + sc] = l;
    }
    // ---- Stage B block (NT x KB); W is [N,K] row-major so rows load like A ----
    #pragma unroll
    for (int r = sr; r < NT; r += 32) {
      const float4 v = *(const float4*)(We + (size_t)r * K + (k0 + sc));
      v4bf h, l;
      h[0] = (__bf16)v.x; l[0] = (__bf16)(v.x - (float)h[0]);
      h[1] = (__bf16)v.y; l[1] = (__bf16)(v.y - (float)h[1]);
      h[2] = (__bf16)v.z; l[2] = (__bf16)(v.z - (float)h[2]);
      h[3] = (__bf16)v.w; l[3] = (__bf16)(v.w - (float)h[3]);
      *(v4bf*)&lBhi[r * KBP + sc] = h;
      *(v4bf*)&lBlo[r * KBP + sc] = l;
    }
    // Touch next K block while we compute on this one (global_prefetch_b8)
    if (k0 + KB < K) {
      __builtin_prefetch(Ae + (size_t)sr * K + (k0 + KB + sc), 0, 0);
      __builtin_prefetch(We + (size_t)(sr & (NT - 1)) * K + (k0 + KB + sc), 0, 0);
    }
    __syncthreads();

    // ---- Build WMMA fragments from LDS ----
    // A (16x32 bf16): lanes 0-15 hold K[0..7],K[16..23]; lanes 16-31 K[8..15],K[24..31]
    v16bf ahi[2], alo[2];
    #pragma unroll
    for (int mt = 0; mt < 2; ++mt) {
      const int row = mWave + mt * 16 + l16;
      const __bf16* bh = &lAhi[row * KBP];
      const __bf16* bl = &lAlo[row * KBP];
      v8bf* ph = (v8bf*)&ahi[mt];
      ph[0] = *(const v8bf*)(bh + half * 8);
      ph[1] = *(const v8bf*)(bh + 16 + half * 8);
      v8bf* pl = (v8bf*)&alo[mt];
      pl[0] = *(const v8bf*)(bl + half * 8);
      pl[1] = *(const v8bf*)(bl + 16 + half * 8);
    }
    // B (32x16 bf16): lane = column n; lanes 0-15 hold K[0..7],K[16..23]; lanes 16-31 K[8..15],K[24..31]
    v16bf bhi[4], blo[4];
    #pragma unroll
    for (int nt = 0; nt < 4; ++nt) {
      const int row = nt * 16 + l16;
      const __bf16* bh = &lBhi[row * KBP + half * 8];
      const __bf16* bl = &lBlo[row * KBP + half * 8];
      v8bf* ph = (v8bf*)&bhi[nt];
      ph[0] = *(const v8bf*)(bh);
      ph[1] = *(const v8bf*)(bh + 16);
      v8bf* pl = (v8bf*)&blo[nt];
      pl[0] = *(const v8bf*)(bl);
      pl[1] = *(const v8bf*)(bl + 16);
    }

    // ---- 3xBF16 split products, 8 independent accumulators per pass ----
    #pragma unroll
    for (int nt = 0; nt < 4; ++nt)
      #pragma unroll
      for (int mt = 0; mt < 2; ++mt)
        acc[mt][nt] = __builtin_amdgcn_wmma_f32_16x16x32_bf16(
            false, ahi[mt], false, blo[nt], (short)0, acc[mt][nt], false, false);
    #pragma unroll
    for (int nt = 0; nt < 4; ++nt)
      #pragma unroll
      for (int mt = 0; mt < 2; ++mt)
        acc[mt][nt] = __builtin_amdgcn_wmma_f32_16x16x32_bf16(
            false, alo[mt], false, bhi[nt], (short)0, acc[mt][nt], false, false);
    #pragma unroll
    for (int nt = 0; nt < 4; ++nt)
      #pragma unroll
      for (int mt = 0; mt < 2; ++mt)
        acc[mt][nt] = __builtin_amdgcn_wmma_f32_16x16x32_bf16(
            false, ahi[mt], false, bhi[nt], (short)0, acc[mt][nt], false, false);
  }

  // ---- Epilogue: C/D layout — VGPR v holds row (v + 8*half), col = lane&15 ----
  #pragma unroll
  for (int mt = 0; mt < 2; ++mt) {
    #pragma unroll
    for (int nt = 0; nt < 4; ++nt) {
      #pragma unroll
      for (int v = 0; v < 8; ++v) {
        const int row = mWave + mt * 16 + v + 8 * half;
        Ce[(size_t)row * N + nt * 16 + l16] = acc[mt][nt][v];
      }
    }
  }
}

extern "C" void kernel_launch(void* const* d_in, const int* in_sizes, int n_in,
                              void* d_out, int out_size, void* d_ws, size_t ws_size,
                              hipStream_t stream) {
  (void)in_sizes; (void)n_in; (void)out_size; (void)d_ws; (void)ws_size;
  const float* X = (const float*)d_in[0];  // [E, M, K]
  const float* W = (const float*)d_in[1];  // [E, N, K]
  float*       O = (float*)d_out;          // [E, M, N]
  dim3 grid(N / NT, E);   // 128 x 16 = 2048 workgroups
  dim3 block(256);        // 8 waves (wave32)
  hipLaunchKernelGGL(grouped_gemm_bf16x3_wmma, grid, block, 0, stream, X, W, O);
}
